// ChunkAwareAttention_21852793602746
// MI455X (gfx1250) — compile-verified, matrix operand order checked by reading, not run on hardware
//
#include <hip/hip_runtime.h>
#include <cstdint>

// ---------------- problem constants ----------------
#define B_   4
#define T_   2048
#define D_   512
#define H_   8
#define DK_  64
#define CHUNK_ 64
#define MROWS_ (B_ * T_)      // 8192
#define LOG2E_ 1.44269504088896340736f

typedef __attribute__((ext_vector_type(16))) __bf16 v16bf;
typedef __attribute__((ext_vector_type(8)))  __bf16 v8bf;
typedef __attribute__((ext_vector_type(8)))  float  v8f;

// ---------------- bf16 helpers ----------------
__device__ __forceinline__ __bf16 f2bf(float f) {
    unsigned u = __builtin_bit_cast(unsigned, f);
    unsigned r = (u + 0x7FFFu + ((u >> 16) & 1u)) >> 16;
    unsigned short s = (unsigned short)r;
    return __builtin_bit_cast(__bf16, s);
}
__device__ __forceinline__ float bf2f(__bf16 h) {
    unsigned short s = __builtin_bit_cast(unsigned short, h);
    unsigned u = ((unsigned)s) << 16;
    return __builtin_bit_cast(float, u);
}

// ---------------- WMMA wrapper ----------------
__device__ __forceinline__ v8f wmma_bf16(v16bf a, v16bf b, v8f c) {
    // 8 args: (neg_a, A, neg_b, B, c_mod, C, reuse_a, reuse_b)
    return __builtin_amdgcn_wmma_f32_16x16x32_bf16(false, a, false, b, (short)0, c,
                                                   false, false);
}

// 16x32 bf16 fragment load from a row-major tile (A and B operands):
// lane<16  -> row = lane,    K elements {0..7, 16..23}
// lane>=16 -> row = lane-16, K elements {8..15, 24..31}
__device__ __forceinline__ v16bf load_frag16(const __bf16* base, int ld, int lane) {
    int r    = lane & 15;
    int koff = (lane < 16) ? 0 : 8;
    const __bf16* p = base + (size_t)r * ld + koff;
    union { v16bf v; struct { v8bf lo; v8bf hi; } h; } u;
    u.h.lo = *(const v8bf*)(p);
    u.h.hi = *(const v8bf*)(p + 16);
    return u.v;
}

// ================= weight packing =================
// Pack W [512,512] fp32 row-major (k,n) into bf16 B-fragment layout:
// block = kt*32 + ntile; within block: lane*16 + j (32B contiguous per lane).
__global__ __launch_bounds__(256) void pack_w_kernel(const float* __restrict__ W,
                                                     __bf16* __restrict__ out) {
    int idx = blockIdx.x * 256 + threadIdx.x;       // 512*512 = 262144 total
    int blk    = idx >> 9;
    int within = idx & 511;
    int lane   = within >> 4;
    int j      = within & 15;
    int kt     = blk >> 5;
    int ntile  = blk & 31;
    int n, k;
    if (lane < 16) { n = ntile * 16 + lane;        k = kt * 32 + ((j < 8) ? j : j + 8);  }
    else           { n = ntile * 16 + (lane - 16); k = kt * 32 + ((j < 8) ? j + 8 : j + 16); }
    out[idx] = f2bf(W[(size_t)k * D_ + n]);
}

// ================= fp32 -> bf16 elementwise =================
__global__ __launch_bounds__(256) void cvt_bf16_kernel(const float* __restrict__ in,
                                                       __bf16* __restrict__ out, int n) {
    int i = blockIdx.x * 256 + threadIdx.x;
    if (i < n) out[i] = f2bf(in[i]);
}

// ================= LayerNorm -> bf16 =================
__global__ __launch_bounds__(256) void ln_kernel(const float* __restrict__ x,
                                                 const float* __restrict__ w,
                                                 const float* __restrict__ bvec,
                                                 __bf16* __restrict__ xn) {
    __shared__ float red[256];
    int row = blockIdx.x;
    int tid = threadIdx.x;
    const float* xr = x + (size_t)row * D_;
    float v0 = xr[tid];
    float v1 = xr[tid + 256];

    red[tid] = v0 + v1;
    __syncthreads();
    for (int s = 128; s > 0; s >>= 1) {
        if (tid < s) red[tid] += red[tid + s];
        __syncthreads();
    }
    float mu = red[0] * (1.0f / D_);
    __syncthreads();

    float d0 = v0 - mu, d1 = v1 - mu;
    red[tid] = d0 * d0 + d1 * d1;
    __syncthreads();
    for (int s = 128; s > 0; s >>= 1) {
        if (tid < s) red[tid] += red[tid + s];
        __syncthreads();
    }
    float var = red[0] * (1.0f / D_);
    float rst = rsqrtf(var + 1e-5f);

    xn[(size_t)row * D_ + tid]       = f2bf(d0 * rst * w[tid]       + bvec[tid]);
    xn[(size_t)row * D_ + tid + 256] = f2bf(d1 * rst * w[tid + 256] + bvec[tid + 256]);
}

// ================= WMMA GEMM: A[M,512](bf16) @ Wpacked + bias, then *oscale ===
// MODE 0: out = bf16 row-major [M,512]
// MODE 1: out = fp32 row-major [M,512]
// MODE 2: out = bf16 transposed-per-head [B,H,DK,T] (for V^T)
template <int MODE>
__global__ __launch_bounds__(256) void gemm_kernel(const __bf16* __restrict__ A,
                                                   const __bf16* __restrict__ Wp,
                                                   const float* __restrict__ bias,
                                                   void* __restrict__ out,
                                                   float oscale) {
    int wave = threadIdx.x >> 5;
    int lane = threadIdx.x & 31;
    int m0 = blockIdx.x * 64 + (wave & 3) * 16;
    int n0 = blockIdx.y * 128 + (wave >> 2) * 64;

    v8f acc[4];
#pragma unroll
    for (int nt = 0; nt < 4; ++nt) acc[nt] = (v8f){0,0,0,0,0,0,0,0};

    for (int kt = 0; kt < 16; ++kt) {
        v16bf a = load_frag16(A + (size_t)m0 * D_ + kt * 32, D_, lane);
#pragma unroll
        for (int nt = 0; nt < 4; ++nt) {
            int ntile = (n0 >> 4) + nt;
            v16bf bfr = *(const v16bf*)(Wp + ((size_t)(kt * 32 + ntile)) * 512 + lane * 16);
            acc[nt] = wmma_bf16(a, bfr, acc[nt]);
        }
    }

#pragma unroll
    for (int nt = 0; nt < 4; ++nt) {
#pragma unroll
        for (int r = 0; r < 8; ++r) {
            int n = n0 + nt * 16 + (lane & 15);
            int m = m0 + r + ((lane < 16) ? 0 : 8);
            float v = (acc[nt][r] + (bias ? bias[n] : 0.0f)) * oscale;
            if (MODE == 0) {
                ((__bf16*)out)[(size_t)m * D_ + n] = f2bf(v);
            } else if (MODE == 1) {
                ((float*)out)[(size_t)m * D_ + n] = v;
            } else {
                int t  = m & (T_ - 1);
                int bb = m >> 11;           // m / T_
                int hh = n >> 6;            // n / DK_
                int dk = n & 63;
                ((__bf16*)out)[(((size_t)(bb * H_ + hh)) * DK_ + dk) * T_ + t] = f2bf(v);
            }
        }
    }
}

// ================= K' = K + pk (broadcast over batch) =================
__global__ __launch_bounds__(256) void addpk_kernel(__bf16* __restrict__ Kp,
                                                    const __bf16* __restrict__ pk) {
    int i = blockIdx.x * 256 + threadIdx.x;          // B*T*D = 4194304
    int j = i & (T_ * D_ - 1);                       // t*D + d
    Kp[i] = f2bf(bf2f(Kp[i]) + bf2f(pk[j]));
}

// ================= flash attention (block-causal, 64-chunks) =================
// grid = (32 qchunks, H, B), block = 128 (4 waves). Each wave: one 16-query tile.
// Q is pre-scaled by 1/sqrt(DK) in its GEMM epilogue; 64 keys per loop step.
__global__ __launch_bounds__(128) void attn_kernel(const __bf16* __restrict__ Q,
                                                   const __bf16* __restrict__ Kp,
                                                   const __bf16* __restrict__ VT,
                                                   __bf16* __restrict__ AO) {
    __shared__ __bf16 pbuf[4][16 * 64];   // per-wave P tile (16 queries x 64 keys)
    int wave = threadIdx.x >> 5;
    int lane = threadIdx.x & 31;
    int qc = blockIdx.x, h = blockIdx.y, b = blockIdx.z;
    int qt = qc * CHUNK_ + wave * 16;

    const __bf16* qbase = Q + ((size_t)(b * T_ + qt)) * D_ + h * DK_;
    v16bf aq0 = load_frag16(qbase,      D_, lane);   // K-dim 0..31
    v16bf aq1 = load_frag16(qbase + 32, D_, lane);   // K-dim 32..63

    float m[8], l[8];
    v8f acc[4];
#pragma unroll
    for (int r = 0; r < 8; ++r) { m[r] = -INFINITY; l[r] = 0.0f; }
#pragma unroll
    for (int nt = 0; nt < 4; ++nt) acc[nt] = (v8f){0,0,0,0,0,0,0,0};

    int kend = (qc + 1) * CHUNK_;         // keys [0, kend), multiple of 64

    const __bf16* kbase = Kp + ((size_t)b * T_) * D_ + h * DK_;
    const __bf16* vbase = VT + ((size_t)(b * H_ + h)) * DK_ * T_;

    for (int kv = 0; kv < kend; kv += 64) {
        // prefetch next key chunk (K' rows + V^T columns) into cache hierarchy
        if (kv + 64 < kend) {
            __builtin_prefetch(kbase + (size_t)(kv + 64 + lane * 2) * D_, 0, 3);
            __builtin_prefetch(vbase + (size_t)(lane * 2) * T_ + kv + 64, 0, 3);
        }

        // ---- S = Q K'^T for 64 keys (four 16x16 tiles), fp32 accum ----
        v8f s[4];
#pragma unroll
        for (int j = 0; j < 4; ++j) {
            const __bf16* kb = kbase + (size_t)(kv + j * 16) * D_;
            v8f sj = (v8f){0,0,0,0,0,0,0,0};
            sj = wmma_bf16(aq0, load_frag16(kb,      D_, lane), sj);
            sj = wmma_bf16(aq1, load_frag16(kb + 32, D_, lane), sj);
            s[j] = sj;
        }

        // ---- online softmax: one update per 64 keys ----
#pragma unroll
        for (int r = 0; r < 8; ++r) {
            float v0 = s[0][r], v1 = s[1][r], v2 = s[2][r], v3 = s[3][r];
            float mx = fmaxf(fmaxf(v0, v1), fmaxf(v2, v3));
            mx = fmaxf(mx, __shfl_xor(mx, 1));
            mx = fmaxf(mx, __shfl_xor(mx, 2));
            mx = fmaxf(mx, __shfl_xor(mx, 4));
            mx = fmaxf(mx, __shfl_xor(mx, 8));
            float mn = fmaxf(m[r], mx);
            float p0 = exp2f((v0 - mn) * LOG2E_);
            float p1 = exp2f((v1 - mn) * LOG2E_);
            float p2 = exp2f((v2 - mn) * LOG2E_);
            float p3 = exp2f((v3 - mn) * LOG2E_);
            float rs = (p0 + p1) + (p2 + p3);
            rs += __shfl_xor(rs, 1);
            rs += __shfl_xor(rs, 2);
            rs += __shfl_xor(rs, 4);
            rs += __shfl_xor(rs, 8);
            float alpha = exp2f((m[r] - mn) * LOG2E_);
            l[r] = l[r] * alpha + rs;
            m[r] = mn;
#pragma unroll
            for (int nt = 0; nt < 4; ++nt) acc[nt][r] *= alpha;
            // stash P (C-layout -> LDS, row-major 16x64)
            int row = r + ((lane < 16) ? 0 : 8);
            int col = lane & 15;
            pbuf[wave][row * 64 + col]      = f2bf(p0);
            pbuf[wave][row * 64 + 16 + col] = f2bf(p1);
            pbuf[wave][row * 64 + 32 + col] = f2bf(p2);
            pbuf[wave][row * 64 + 48 + col] = f2bf(p3);
        }

        // ---- O += P @ V  (two A-frags from LDS, B-frags from V^T) ----
        v16bf pa0 = load_frag16(&pbuf[wave][0],  64, lane);   // keys kv..kv+31
        v16bf pa1 = load_frag16(&pbuf[wave][32], 64, lane);   // keys kv+32..kv+63
#pragma unroll
        for (int nt = 0; nt < 4; ++nt) {
            const __bf16* vb = vbase + (size_t)(nt * 16) * T_ + kv;
            acc[nt] = wmma_bf16(pa0, load_frag16(vb,      T_, lane), acc[nt]);
            acc[nt] = wmma_bf16(pa1, load_frag16(vb + 32, T_, lane), acc[nt]);
        }
    }

    // ---- normalize and write [B,T,D] bf16 ----
#pragma unroll
    for (int nt = 0; nt < 4; ++nt) {
#pragma unroll
        for (int r = 0; r < 8; ++r) {
            int row = r + ((lane < 16) ? 0 : 8);
            int t = qt + row;
            int n = h * DK_ + nt * 16 + (lane & 15);
            AO[((size_t)(b * T_ + t)) * D_ + n] = f2bf(acc[nt][r] / l[r]);
        }
    }
}

// ================= host launcher =================
extern "C" void kernel_launch(void* const* d_in, const int* in_sizes, int n_in,
                              void* d_out, int out_size, void* d_ws, size_t ws_size,
                              hipStream_t stream) {
    (void)in_sizes; (void)n_in; (void)out_size; (void)ws_size;
    const float* x       = (const float*)d_in[0];
    const float* pos_enc = (const float*)d_in[1];
    // d_in[2] = mask: recomputed analytically (block-causal at CHUNK=64)
    const float* ln_w = (const float*)d_in[3];
    const float* ln_b = (const float*)d_in[4];
    const float* Wq = (const float*)d_in[5];  const float* bq = (const float*)d_in[6];
    const float* Wk = (const float*)d_in[7];  const float* bk = (const float*)d_in[8];
    const float* Wv = (const float*)d_in[9];  const float* bv = (const float*)d_in[10];
    const float* Wpos = (const float*)d_in[11];
    const float* Wout = (const float*)d_in[12];
    const float* bout = (const float*)d_in[13];
    float* out = (float*)d_out;

    // workspace carve-up (256B aligned)
    char* wsp = (char*)d_ws;
    auto carve = [&](size_t bytes) {
        char* p = wsp;
        wsp += (bytes + 255) & ~(size_t)255;
        return p;
    };
    const size_t WPACK = (size_t)512 * 512 * sizeof(__bf16);
    __bf16* xn    = (__bf16*)carve((size_t)MROWS_ * D_ * 2);
    __bf16* posb  = (__bf16*)carve((size_t)T_ * D_ * 2);
    __bf16* WqP   = (__bf16*)carve(WPACK);
    __bf16* WkP   = (__bf16*)carve(WPACK);
    __bf16* WvP   = (__bf16*)carve(WPACK);
    __bf16* WposP = (__bf16*)carve(WPACK);
    __bf16* WoutP = (__bf16*)carve(WPACK);
    __bf16* Qb    = (__bf16*)carve((size_t)MROWS_ * D_ * 2);
    __bf16* Kpb   = (__bf16*)carve((size_t)MROWS_ * D_ * 2);
    __bf16* VTb   = (__bf16*)carve((size_t)MROWS_ * D_ * 2);
    __bf16* pkb   = (__bf16*)carve((size_t)T_ * D_ * 2);
    __bf16* AOb   = (__bf16*)carve((size_t)MROWS_ * D_ * 2);

    // 1) pack weights into WMMA B-fragment layout
    pack_w_kernel<<<1024, 256, 0, stream>>>(Wq,   WqP);
    pack_w_kernel<<<1024, 256, 0, stream>>>(Wk,   WkP);
    pack_w_kernel<<<1024, 256, 0, stream>>>(Wv,   WvP);
    pack_w_kernel<<<1024, 256, 0, stream>>>(Wpos, WposP);
    pack_w_kernel<<<1024, 256, 0, stream>>>(Wout, WoutP);

    // 2) pos_enc -> bf16 ; LayerNorm(x) -> bf16
    cvt_bf16_kernel<<<(T_ * D_) / 256, 256, 0, stream>>>(pos_enc, posb, T_ * D_);
    ln_kernel<<<MROWS_, 256, 0, stream>>>(x, ln_w, ln_b, xn);

    // 3) QKV + positional GEMMs. Q is pre-scaled by 1/sqrt(DK) (exact pow2).
    dim3 g8k(MROWS_ / 64, 4), g2k(T_ / 64, 4);
    gemm_kernel<0><<<g8k, 256, 0, stream>>>(xn,   WqP,   bq,      (void*)Qb,  0.125f);
    gemm_kernel<0><<<g8k, 256, 0, stream>>>(xn,   WkP,   bk,      (void*)Kpb, 1.0f);
    gemm_kernel<2><<<g8k, 256, 0, stream>>>(xn,   WvP,   bv,      (void*)VTb, 1.0f);
    gemm_kernel<0><<<g2k, 256, 0, stream>>>(posb, WposP, nullptr, (void*)pkb, 1.0f);

    // 4) K' = K + pk  (folds q·pk^T into q·K'^T)
    addpk_kernel<<<(MROWS_ * D_) / 256, 256, 0, stream>>>(Kpb, pkb);

    // 5) block-causal flash attention
    attn_kernel<<<dim3(T_ / CHUNK_, H_, B_), 128, 0, stream>>>(Qb, Kpb, VTb, AOb);

    // 6) output projection -> fp32 d_out
    gemm_kernel<1><<<g8k, 256, 0, stream>>>(AOb, WoutP, bout, (void*)out, 1.0f);
}
